// Encoder_54838142435519
// MI455X (gfx1250) — matrix operand order repaired and checked by reading, not compiled, not run on hardware
//
#include <hip/hip_runtime.h>
#include <cstddef>
#include <cstdint>

// ---------------------------------------------------------------------------
// Types for CDNA5 WMMA (wave32, 16x16x32 f16 -> f32 accumulate)
// ---------------------------------------------------------------------------
typedef _Float16 h16;
typedef __attribute__((ext_vector_type(16))) _Float16 v16h;
typedef __attribute__((ext_vector_type(8)))  float    v8f;

union U16 { v16h v; h16 e[16]; };
union UC  { v8f  v; float e[8]; };

static __device__ __forceinline__ v8f wmma_f16(v16h a, v16h b, v8f c) {
    // (neg_a, A, neg_b, B, c_mod, C, reuse_a, reuse_b)
    return __builtin_amdgcn_wmma_f32_16x16x32_f16(false, a, false, b,
                                                  (short)0, c, false, false);
}

// CDNA5 async global->LDS copy (16B per lane), tracked by ASYNCcnt.
// LDS destination = low 32 bits of the generic pointer (flat-aperture rule).
static __device__ __forceinline__ void async_copy_b128(const h16* gsrc, h16* ldst) {
    unsigned           lds = (unsigned)(uintptr_t)ldst;
    unsigned long long ga  = (unsigned long long)(uintptr_t)gsrc;
    asm volatile("global_load_async_to_lds_b128 %0, %1, off"
                 :: "v"(lds), "v"(ga) : "memory");
}
static __device__ __forceinline__ void async_wait0() {
    asm volatile("s_wait_asynccnt 0x0" ::: "memory");
}

// ---------------------------------------------------------------------------
// Model constants
// ---------------------------------------------------------------------------
#define N_B     4
#define N_S     2048
#define N_D     512
#define N_H     8
#define N_DK    64
#define N_FF    2048
#define N_L     4
#define N_BS    (N_B * N_S)   // 8192 tokens

// ---------------------------------------------------------------------------
// f32 -> f16 weight conversion
// ---------------------------------------------------------------------------
__global__ void cvt_f16_kernel(const float* __restrict__ in, h16* __restrict__ out, long n) {
    long i = (long)blockIdx.x * blockDim.x + threadIdx.x;
    if (i < n) out[i] = (h16)in[i];
}

// ---------------------------------------------------------------------------
// Embedding: x = tok_embed[ids] + pe  (f32 residual + f16 activation copy)
// ---------------------------------------------------------------------------
__global__ void embed_kernel(const int* __restrict__ ids,
                             const float* __restrict__ emb,
                             const float* __restrict__ pe,
                             float* __restrict__ x, h16* __restrict__ xh) {
    int t = blockIdx.x;            // token index in [0, B*S)
    int s = t % N_S;
    int v = ids[t];
    for (int d = threadIdx.x; d < N_D; d += blockDim.x) {
        float val = emb[v * N_D + d] + pe[s * N_D + d];
        x[(size_t)t * N_D + d]  = val;
        xh[(size_t)t * N_D + d] = (h16)val;
    }
}

// ---------------------------------------------------------------------------
// LayerNorm over D=512 (one block per token), writes f32 + f16
// ---------------------------------------------------------------------------
__global__ __launch_bounds__(256)
void ln_kernel(const float* __restrict__ y, const float* __restrict__ g,
               const float* __restrict__ b, float* __restrict__ x,
               h16* __restrict__ xh) {
    size_t t = blockIdx.x;
    int tid = threadIdx.x;
    float v0 = y[t * N_D + tid];
    float v1 = y[t * N_D + 256 + tid];
    __shared__ float s1[256], s2[256];
    s1[tid] = v0 + v1;
    s2[tid] = v0 * v0 + v1 * v1;
    __syncthreads();
    for (int off = 128; off > 0; off >>= 1) {
        if (tid < off) { s1[tid] += s1[tid + off]; s2[tid] += s2[tid + off]; }
        __syncthreads();
    }
    float mu  = s1[0] * (1.0f / N_D);
    float var = s2[0] * (1.0f / N_D) - mu * mu;
    float inv = rsqrtf(var + 1e-5f);
    float o0 = (v0 - mu) * inv * g[tid] + b[tid];
    float o1 = (v1 - mu) * inv * g[256 + tid] + b[256 + tid];
    x[t * N_D + tid]        = o0;  xh[t * N_D + tid]        = (h16)o0;
    x[t * N_D + 256 + tid]  = o1;  xh[t * N_D + 256 + tid]  = (h16)o1;
}

__global__ void copy_kernel(const float* __restrict__ a, float* __restrict__ o, long n) {
    long i = (long)blockIdx.x * blockDim.x + threadIdx.x;
    if (i < n) o[i] = a[i];
}

// ---------------------------------------------------------------------------
// Tiled WMMA GEMM: C[M,N] = A[M,K] * B[K,N]   (A,B f16 row-major, acc f32)
// Block tile 128x64, 8 waves (4x2), each wave 32x32 = 2x2 WMMA tiles.
// Double-buffered LDS; A tiles staged with global_load_async_to_lds_b128
// (ASYNCcnt), B tiles staged synchronously with a transpose scatter.
// Epilogue: optional residual add (f32), optional ReLU, write f32 and/or f16.
// ---------------------------------------------------------------------------
__global__ __launch_bounds__(256)
void gemm_kernel(const h16* __restrict__ A, const h16* __restrict__ B,
                 const float* __restrict__ res, float* __restrict__ outF,
                 h16* __restrict__ outH, int M, int N, int K, int relu) {
    __shared__ h16 As[2][128 * 32];   // [row][k], matches A-frag layout
    __shared__ h16 Bt[2][64 * 32];    // transposed [col][k] -> contiguous B-frags

    const int tid  = threadIdx.x;
    const int lane = tid & 31;
    const int w    = tid >> 5;          // 0..7
    const int wm   = w & 3;             // 4 wave-rows of 32
    const int wn   = w >> 2;            // 2 wave-cols of 32
    const int bm   = blockIdx.y * 128;
    const int bn   = blockIdx.x * 64;

    // stage A tile (async DMA into LDS, 2 x 16B chunks per thread)
    auto stage_a = [&](h16* dst, int k0) {
#pragma unroll
        for (int c = tid; c < 512; c += 256) {
            int r  = c >> 2;
            int cb = (c & 3) * 8;
            async_copy_b128(A + (size_t)(bm + r) * K + k0 + cb, dst + r * 32 + cb);
        }
    };
    // stage B tile transposed (sync, 8 scalar LDS stores per thread)
    auto stage_b = [&](h16* dst, int k0) {
        int krow = tid >> 3;
        int cb   = (tid & 7) * 8;
        float4 t4 = *(const float4*)(B + (size_t)(k0 + krow) * N + bn + cb);
        const h16* tp = (const h16*)&t4;
#pragma unroll
        for (int j = 0; j < 8; ++j) dst[(cb + j) * 32 + krow] = tp[j];
    };

    UC acc[2][2];
#pragma unroll
    for (int i = 0; i < 2; ++i)
#pragma unroll
        for (int j = 0; j < 2; ++j)
#pragma unroll
            for (int r = 0; r < 8; ++r) acc[i][j].e[r] = 0.0f;

    const int T = K >> 5;               // number of 32-wide K tiles
    stage_a(As[0], 0);
    stage_b(Bt[0], 0);

    for (int it = 0; it < T; ++it) {
        async_wait0();                   // own async writes to LDS landed
        __syncthreads();                 // everyone's writes visible / reads done
        const int cur = it & 1;

        if (it + 1 < T) {                // prefetch next tile into other buffer
            stage_a(As[cur ^ 1], (it + 1) * 32);
            stage_b(Bt[cur ^ 1], (it + 1) * 32);
            __builtin_prefetch(B + (size_t)((it + 2) * 32) * N + bn, 0, 3);
        }

        // ---- fragments ----
        // A-frag (16x32 f16): lanes 0-15 -> rows M0..15 with K {0..7,16..23};
        // lanes 16-31 -> same rows with K {8..15,24..31}.
        U16 af[2], bf[2];
        const int kb = (lane >> 4) * 8;
#pragma unroll
        for (int tm = 0; tm < 2; ++tm) {
            int r = wm * 32 + tm * 16 + (lane & 15);
            *(float4*)&af[tm].e[0] = *(float4*)(As[cur] + r * 32 + kb);
            *(float4*)&af[tm].e[8] = *(float4*)(As[cur] + r * 32 + kb + 16);
        }
        // B-frag (32x16 f16): lane -> column, K {0..7,16..23} / {8..15,24..31} per lane half
        const int kk = (lane >> 4) * 8;
#pragma unroll
        for (int tn = 0; tn < 2; ++tn) {
            int col = wn * 32 + tn * 16 + (lane & 15);
            *(float4*)&bf[tn].e[0] = *(float4*)(Bt[cur] + col * 32 + kk);
            *(float4*)&bf[tn].e[8] = *(float4*)(Bt[cur] + col * 32 + kk + 16);
        }
#pragma unroll
        for (int tm = 0; tm < 2; ++tm)
#pragma unroll
            for (int tn = 0; tn < 2; ++tn)
                acc[tm][tn].v = wmma_f16(af[tm].v, bf[tn].v, acc[tm][tn].v);
    }

    // ---- epilogue: C layout lanes 0-15 -> rows 0..7, lanes 16-31 -> rows 8..15
#pragma unroll
    for (int tm = 0; tm < 2; ++tm)
#pragma unroll
        for (int tn = 0; tn < 2; ++tn) {
            int rbase = bm + wm * 32 + tm * 16 + ((lane < 16) ? 0 : 8);
            int col   = bn + wn * 32 + tn * 16 + (lane & 15);
#pragma unroll
            for (int r = 0; r < 8; ++r) {
                size_t idx = (size_t)(rbase + r) * N + col;
                float v = acc[tm][tn].e[r];
                if (res)  v += res[idx];
                if (relu) v = fmaxf(v, 0.0f);
                if (outF) outF[idx] = v;
                if (outH) outH[idx] = (h16)v;
            }
        }
}

// ---------------------------------------------------------------------------
// Flash-style attention: one wave per (b, h, 16-query tile).
// K/V blocks staged with async global->LDS DMA; S = Q*K^T (4 WMMA / 32-key
// block), online softmax, ctx += P*V (4 WMMA).
// ---------------------------------------------------------------------------
__global__ __launch_bounds__(32)
void attn_kernel(const h16* __restrict__ qh, const h16* __restrict__ kh,
                 const h16* __restrict__ vh, const int* __restrict__ mask,
                 h16* __restrict__ ctxh) {
    const int q0   = blockIdx.x * 16;
    const int h    = blockIdx.y;
    const int b    = blockIdx.z;
    const int lane = threadIdx.x;

    __shared__ h16 Kl[32 * 64];
    __shared__ h16 Vl[32 * 64];
    __shared__ h16 Pl[16 * 32];

    // Q fragments (A layout), two 32-wide d-chunks of the 64-wide head dim
    U16 aq[2];
    {
        int row = q0 + (lane & 15);
        int kb  = (lane >> 4) * 8;
        const h16* qrow = qh + ((size_t)(b * N_S + row)) * N_D + h * N_DK;
#pragma unroll
        for (int dc = 0; dc < 2; ++dc) {
#pragma unroll
            for (int i = 0; i < 8; ++i) {
                aq[dc].e[i]     = qrow[dc * 32 + kb + i];
                aq[dc].e[8 + i] = qrow[dc * 32 + kb + 16 + i];
            }
        }
    }

    float m[8], lsum[8];
    UC ctx[4];
#pragma unroll
    for (int r = 0; r < 8; ++r) { m[r] = -1e30f; lsum[r] = 0.0f; }
#pragma unroll
    for (int tn = 0; tn < 4; ++tn)
#pragma unroll
        for (int r = 0; r < 8; ++r) ctx[tn].e[r] = 0.0f;

    for (int kb0 = 0; kb0 < N_S; kb0 += 32) {
        __syncthreads();
        // stage K,V block (32 keys x 64 dims) via async DMA; one row per lane
        {
            const h16* ks = kh + ((size_t)(b * N_S + kb0 + lane)) * N_D + h * N_DK;
            const h16* vs = vh + ((size_t)(b * N_S + kb0 + lane)) * N_D + h * N_DK;
#pragma unroll
            for (int c = 0; c < 64; c += 8) {
                async_copy_b128(ks + c, Kl + lane * 64 + c);
                async_copy_b128(vs + c, Vl + lane * 64 + c);
            }
        }
        async_wait0();
        __syncthreads();

        // S = Q * K^T   (two 16-key tiles, contraction over d split in two)
        UC sc[2];
#pragma unroll
        for (int kt = 0; kt < 2; ++kt)
#pragma unroll
            for (int r = 0; r < 8; ++r) sc[kt].e[r] = 0.0f;

#pragma unroll
        for (int kt = 0; kt < 2; ++kt) {
            int key = kt * 16 + (lane & 15);
            int kks = (lane >> 4) * 8;
#pragma unroll
            for (int dc = 0; dc < 2; ++dc) {
                U16 bk;   // B-frag of K^T: lane=key column, d's {0..7,16..23} / {8..15,24..31}
                *(float4*)&bk.e[0] = *(float4*)(Kl + key * 64 + dc * 32 + kks);
                *(float4*)&bk.e[8] = *(float4*)(Kl + key * 64 + dc * 32 + kks + 16);
                sc[kt].v = wmma_f16(aq[dc].v, bk.v, sc[kt].v);
            }
        }

        // scale + pad mask (key-dependent only)
#pragma unroll
        for (int kt = 0; kt < 2; ++kt) {
            int key = kb0 + kt * 16 + (lane & 15);
            int mk  = mask[b * N_S + key];
#pragma unroll
            for (int r = 0; r < 8; ++r) {
                float s = sc[kt].e[r] * 0.125f;   // 1/sqrt(64)
                sc[kt].e[r] = (mk == 0) ? -1e9f : s;
            }
        }

        // online softmax per query row (rows live per-VGPR, keys across 16 lanes)
#pragma unroll
        for (int r = 0; r < 8; ++r) {
            float bm_ = fmaxf(sc[0].e[r], sc[1].e[r]);
#pragma unroll
            for (int off = 8; off >= 1; off >>= 1)
                bm_ = fmaxf(bm_, __shfl_xor(bm_, off, 16));
            float newm  = fmaxf(m[r], bm_);
            float alpha = __expf(m[r] - newm);
            float p0 = __expf(sc[0].e[r] - newm);
            float p1 = __expf(sc[1].e[r] - newm);
            float ps = p0 + p1;
#pragma unroll
            for (int off = 8; off >= 1; off >>= 1)
                ps += __shfl_xor(ps, off, 16);
            lsum[r] = lsum[r] * alpha + ps;
            m[r] = newm;
#pragma unroll
            for (int tn = 0; tn < 4; ++tn) ctx[tn].e[r] *= alpha;
            sc[0].e[r] = p0;
            sc[1].e[r] = p1;
        }

        // C-layout -> A-layout transpose of P through LDS
        __syncthreads();
#pragma unroll
        for (int kt = 0; kt < 2; ++kt)
#pragma unroll
            for (int r = 0; r < 8; ++r) {
                int prow = (lane < 16) ? r : (r + 8);
                Pl[prow * 32 + kt * 16 + (lane & 15)] = (h16)sc[kt].e[r];
            }
        __syncthreads();

        U16 ap;
        {
            int prow = lane & 15;
            int pkb  = (lane >> 4) * 16;
            *(float4*)&ap.e[0] = *(float4*)(Pl + prow * 32 + pkb);
            *(float4*)&ap.e[8] = *(float4*)(Pl + prow * 32 + pkb + 8);
        }
        // ctx += P * V  (contraction over 32 keys; 4 d-tiles of 16)
        {
            int kks = (lane >> 4) * 16;
#pragma unroll
            for (int tn = 0; tn < 4; ++tn) {
                U16 bv;
                int dcol = tn * 16 + (lane & 15);
#pragma unroll
                for (int i = 0; i < 16; ++i)
                    bv.e[i] = Vl[(kks + i) * 64 + dcol];
                ctx[tn].v = wmma_f16(ap.v, bv.v, ctx[tn].v);
            }
        }
    }

    // normalize and write ctx (f16, head-interleaved [B,S,H*64])
#pragma unroll
    for (int tn = 0; tn < 4; ++tn)
#pragma unroll
        for (int r = 0; r < 8; ++r) {
            int orow = q0 + ((lane < 16) ? r : (8 + r));
            int ocol = h * N_DK + tn * 16 + (lane & 15);
            float denom = (lsum[r] > 0.0f) ? lsum[r] : 1.0f;
            ctxh[((size_t)(b * N_S + orow)) * N_D + ocol] = (h16)(ctx[tn].e[r] / denom);
        }
}

// ---------------------------------------------------------------------------
// Host-side orchestration
// ---------------------------------------------------------------------------
extern "C" void kernel_launch(void* const* d_in, const int* in_sizes, int n_in,
                              void* d_out, int out_size, void* d_ws, size_t ws_size,
                              hipStream_t stream) {
    const int*   ids    = (const int*)  d_in[0];
    const int*   amask  = (const int*)  d_in[1];
    const float* emb    = (const float*)d_in[2];
    const float* pe     = (const float*)d_in[3];
    const float* Wq     = (const float*)d_in[4];
    const float* Wk     = (const float*)d_in[5];
    const float* Wv     = (const float*)d_in[6];
    const float* Wo     = (const float*)d_in[7];
    const float* ln1g   = (const float*)d_in[8];
    const float* ln1b   = (const float*)d_in[9];
    const float* W1     = (const float*)d_in[10];
    const float* W2     = (const float*)d_in[11];
    const float* ln2g   = (const float*)d_in[12];
    const float* ln2b   = (const float*)d_in[13];

    char* wsb = (char*)d_ws;
    size_t off = 0;
    auto take = [&](size_t bytes) -> char* {
        char* p = wsb + off;
        off += (bytes + 255) & ~(size_t)255;
        return p;
    };

    const size_t szW  = (size_t)N_L * N_D * N_D;    // per proj-weight elements
    const size_t szFF = (size_t)N_L * N_D * N_FF;

    h16* WQ16 = (h16*)take(szW  * sizeof(h16));
    h16* WK16 = (h16*)take(szW  * sizeof(h16));
    h16* WV16 = (h16*)take(szW  * sizeof(h16));
    h16* WO16 = (h16*)take(szW  * sizeof(h16));
    h16* W116 = (h16*)take(szFF * sizeof(h16));
    h16* W216 = (h16*)take(szFF * sizeof(h16));
    float* xf = (float*)take((size_t)N_BS * N_D * sizeof(float));
    float* yf = (float*)take((size_t)N_BS * N_D * sizeof(float));
    h16* xh   = (h16*)take((size_t)N_BS * N_D * sizeof(h16));
    h16* qh   = (h16*)take((size_t)N_BS * N_D * sizeof(h16));
    h16* kh   = (h16*)take((size_t)N_BS * N_D * sizeof(h16));
    h16* vh   = (h16*)take((size_t)N_BS * N_D * sizeof(h16));
    h16* ch   = (h16*)take((size_t)N_BS * N_D * sizeof(h16));
    h16* hh   = (h16*)take((size_t)N_BS * N_FF * sizeof(h16));

    // --- weight conversion to f16 ---
    auto cvt = [&](const float* src, h16* dst, size_t n) {
        long nl = (long)n;
        cvt_f16_kernel<<<(unsigned)((nl + 255) / 256), 256, 0, stream>>>(src, dst, nl);
    };
    cvt(Wq, WQ16, szW); cvt(Wk, WK16, szW); cvt(Wv, WV16, szW); cvt(Wo, WO16, szW);
    cvt(W1, W116, szFF); cvt(W2, W216, szFF);

    // --- embedding ---
    embed_kernel<<<N_BS, 256, 0, stream>>>(ids, emb, pe, xf, xh);

    for (int l = 0; l < N_L; ++l) {
        const h16* wq = WQ16 + (size_t)l * N_D * N_D;
        const h16* wk = WK16 + (size_t)l * N_D * N_D;
        const h16* wv = WV16 + (size_t)l * N_D * N_D;
        const h16* wo = WO16 + (size_t)l * N_D * N_D;
        const h16* w1 = W116 + (size_t)l * N_D * N_FF;
        const h16* w2 = W216 + (size_t)l * N_FF * N_D;

        dim3 gProj(N_D / 64, N_BS / 128);     // (8, 64)
        dim3 gFF1 (N_FF / 64, N_BS / 128);    // (32, 64)

        // Q, K, V projections (f16 outputs only)
        gemm_kernel<<<gProj, 256, 0, stream>>>(xh, wq, nullptr, nullptr, qh,
                                               N_BS, N_D, N_D, 0);
        gemm_kernel<<<gProj, 256, 0, stream>>>(xh, wk, nullptr, nullptr, kh,
                                               N_BS, N_D, N_D, 0);
        gemm_kernel<<<gProj, 256, 0, stream>>>(xh, wv, nullptr, nullptr, vh,
                                               N_BS, N_D, N_D, 0);

        // attention
        dim3 gAttn(N_S / 16, N_H, N_B);
        attn_kernel<<<gAttn, 32, 0, stream>>>(qh, kh, vh, amask, ch);

        // Wo projection + residual -> y, then LN1 -> x (f32 + f16)
        gemm_kernel<<<gProj, 256, 0, stream>>>(ch, wo, xf, yf, nullptr,
                                               N_BS, N_D, N_D, 0);
        ln_kernel<<<N_BS, 256, 0, stream>>>(yf, ln1g + (size_t)l * N_D,
                                            ln1b + (size_t)l * N_D, xf, xh);

        // FFN: h = relu(x@W1) -> f16 ; y = h@W2 + x ; LN2 -> x
        gemm_kernel<<<gFF1, 256, 0, stream>>>(xh, w1, nullptr, nullptr, hh,
                                              N_BS, N_FF, N_D, 1);
        gemm_kernel<<<gProj, 256, 0, stream>>>(hh, w2, xf, yf, nullptr,
                                               N_BS, N_D, N_FF, 0);
        ln_kernel<<<N_BS, 256, 0, stream>>>(yf, ln2g + (size_t)l * N_D,
                                            ln2b + (size_t)l * N_D, xf, xh);
    }

    // final output (f32)
    long n = (long)N_BS * N_D;
    copy_kernel<<<(unsigned)((n + 255) / 256), 256, 0, stream>>>(xf, (float*)d_out, n);
}
